// GlobalEdgeGnn_13477607374970
// MI455X (gfx1250) — matrix operand order repaired and checked by reading, not compiled, not run on hardware
//
#include <hip/hip_runtime.h>
#include <hip/hip_bf16.h>

typedef __attribute__((ext_vector_type(16))) __bf16 v16bf;
typedef __attribute__((ext_vector_type(8)))  float  v8f;

#define NN 50000     // nodes
#define EE 500000    // undirected edges
#define DD 64        // feature dim
#define HW 128       // hidden width

// A-matrix (16-bit, 16x32) K index pattern per VGPR pair v, lower half-lanes.
static __device__ __forceinline__ int kpatA(int v) { return (v < 4) ? (2 * v) : (8 + 2 * v); }

// ---------------- weight transpose + bf16 convert: dst[c][k] = src[k][c] ----------------
__global__ void kprep_w(const float* __restrict__ src, __bf16* __restrict__ dst, int K, int Nc) {
  int i = blockIdx.x * blockDim.x + threadIdx.x;
  if (i >= K * Nc) return;
  int k = i / Nc, c = i % Nc;
  dst[(size_t)c * K + k] = (__bf16)src[i];
}

// ---------------- initial node features: segment sum of incident edge feats + degree ----------------
__global__ void kscatter(const float* __restrict__ ef, const int* __restrict__ srcI,
                         const int* __restrict__ dstI, float* __restrict__ xacc,
                         float* __restrict__ deg) {
  long long i = (long long)blockIdx.x * blockDim.x + threadIdx.x;
  if (i >= 2LL * EE * DD) return;
  int e2 = (int)(i >> 6);
  int c  = (int)(i & 63);
  int e    = (e2 < EE) ? e2 : e2 - EE;
  int node = (e2 < EE) ? dstI[e] : srcI[e];
  atomicAdd(&xacc[(size_t)node * DD + c], ef[(size_t)e * DD + c]);
  if (c == 0) atomicAdd(&deg[node], 1.0f);
}

// ---------------- x = (residual? x : 0) + xacc/max(deg,1); xacc := 0 ----------------
__global__ void knorm(float* __restrict__ x, float* __restrict__ xacc,
                      const float* __restrict__ deg, int residual) {
  int i = blockIdx.x * blockDim.x + threadIdx.x;
  if (i >= NN * DD) return;
  float dg = deg[i >> 6];
  dg = dg < 1.f ? 1.f : dg;
  float v = xacc[i] / dg;
  x[i] = residual ? (x[i] + v) : v;
  xacc[i] = 0.f;
}

// Build the 4 K-step A fragments for one 16-row tile: row = [x[d], x[s]] (concat).
static __device__ __forceinline__ void build_afrag(const float* __restrict__ xd,
                                                   const float* __restrict__ xs,
                                                   int hh, v16bf frag[4]) {
#pragma unroll
  for (int st = 0; st < 4; ++st)
#pragma unroll
    for (int v = 0; v < 8; ++v) {
      int kk = 32 * st + kpatA(v) + 8 * hh;
      const float* p = (kk < DD) ? (xd + kk) : (xs + (kk - DD));
      float2 f = *(const float2*)p;
      frag[st][2 * v]     = (__bf16)f.x;
      frag[st][2 * v + 1] = (__bf16)f.y;
    }
}

// Load one B fragment (K base = 32*st) from transposed bf16 weights.
static __device__ __forceinline__ v16bf load_bfrag(const __bf16* __restrict__ wrow,
                                                   int st, int hh) {
  v16bf bfrag;
#pragma unroll
  for (int v = 0; v < 8; ++v) {
    int kk = 32 * st + kpatA(v) + 8 * hh;
    bfrag[2 * v]     = wrow[kk];
    bfrag[2 * v + 1] = wrow[kk + 1];
  }
  return bfrag;
}

// ---------------- fused NodeConv layer: gather -> GEMM1(relu) -> GEMM2(relu) -> scatter ----------------
// Each wave owns TWO 16-row tiles (32 directed edges); B fragments shared across both.
__global__ __launch_bounds__(128)
void kconv(const float* __restrict__ x, float* __restrict__ xacc,
           const int* __restrict__ srcI, const int* __restrict__ dstI,
           const __bf16* __restrict__ w1T, const float* __restrict__ b1,
           const __bf16* __restrict__ w2T, const float* __restrict__ b2) {
  __shared__ __bf16 hid[4][32][HW];
  __shared__ int dsti[4][32];
  const int wave = threadIdx.x >> 5;
  const int lane = threadIdx.x & 31;
  const int hh   = lane >> 4;    // half-lane select
  const int m16  = lane & 15;
  const int pairTiles = (2 * EE) / 32;
  const int pt = blockIdx.x * 4 + wave;
  const bool active = pt < pairTiles;

  v16bf afA[4], afB[4];
  if (active) {
    // fetch endpoint indices for both tiles
#pragma unroll
    for (int t = 0; t < 2; ++t) {
      int r = pt * 32 + t * 16 + m16;
      int d, s;
      if (r < EE) { d = dstI[r]; s = srcI[r]; }
      else        { d = srcI[r - EE]; s = dstI[r - EE]; }
      if (lane < 16) dsti[wave][t * 16 + lane] = d;
      const float* xd = x + (size_t)d * DD;
      const float* xs = x + (size_t)s * DD;
      if (t == 0) build_afrag(xd, xs, hh, afA);
      else        build_afrag(xd, xs, hh, afB);
    }
    // GEMM1: [32x128] x [128x128] -> relu -> LDS (bf16); B frag shared by both tiles
#pragma unroll
    for (int ct = 0; ct < 8; ++ct) {
      int col = ct * 16 + m16;
      v8f accA = {0.f, 0.f, 0.f, 0.f, 0.f, 0.f, 0.f, 0.f};
      v8f accB = {0.f, 0.f, 0.f, 0.f, 0.f, 0.f, 0.f, 0.f};
      const __bf16* wrow = w1T + (size_t)col * HW;
#pragma unroll
      for (int st = 0; st < 4; ++st) {
        v16bf bfrag = load_bfrag(wrow, st, hh);
        accA = __builtin_amdgcn_wmma_f32_16x16x32_bf16(false, afA[st], false, bfrag,
                                                       (short)0, accA, false, false);
        accB = __builtin_amdgcn_wmma_f32_16x16x32_bf16(false, afB[st], false, bfrag,
                                                       (short)0, accB, false, false);
      }
      float bias = b1[col];
#pragma unroll
      for (int v = 0; v < 8; ++v) {
        float hA = accA[v] + bias; hA = hA > 0.f ? hA : 0.f;
        float hB = accB[v] + bias; hB = hB > 0.f ? hB : 0.f;
        hid[wave][v + 8 * hh][col]      = (__bf16)hA;
        hid[wave][16 + v + 8 * hh][col] = (__bf16)hB;
      }
    }
  }
  __syncthreads();
  if (active) {
    // A' fragments from LDS hidden tiles
    v16bf aA[4], aB[4];
#pragma unroll
    for (int st = 0; st < 4; ++st)
#pragma unroll
      for (int v = 0; v < 8; ++v) {
        int kk = 32 * st + kpatA(v) + 8 * hh;
        aA[st][2 * v] = hid[wave][m16][kk];      aA[st][2 * v + 1] = hid[wave][m16][kk + 1];
        aB[st][2 * v] = hid[wave][16 + m16][kk]; aB[st][2 * v + 1] = hid[wave][16 + m16][kk + 1];
      }
    // GEMM2: [32x128] x [128x64] -> relu -> atomic scatter into xacc
#pragma unroll
    for (int ct = 0; ct < 4; ++ct) {
      int col = ct * 16 + m16;
      v8f accA = {0.f, 0.f, 0.f, 0.f, 0.f, 0.f, 0.f, 0.f};
      v8f accB = {0.f, 0.f, 0.f, 0.f, 0.f, 0.f, 0.f, 0.f};
      const __bf16* wrow = w2T + (size_t)col * HW;
#pragma unroll
      for (int st = 0; st < 4; ++st) {
        v16bf bfrag = load_bfrag(wrow, st, hh);
        accA = __builtin_amdgcn_wmma_f32_16x16x32_bf16(false, aA[st], false, bfrag,
                                                       (short)0, accA, false, false);
        accB = __builtin_amdgcn_wmma_f32_16x16x32_bf16(false, aB[st], false, bfrag,
                                                       (short)0, accB, false, false);
      }
      float bias = b2[col];
#pragma unroll
      for (int v = 0; v < 8; ++v) {
        float hA = accA[v] + bias; hA = hA > 0.f ? hA : 0.f;
        float hB = accB[v] + bias; hB = hB > 0.f ? hB : 0.f;
        int nA = dsti[wave][v + 8 * hh];
        int nB = dsti[wave][16 + v + 8 * hh];
        atomicAdd(&xacc[(size_t)nA * DD + col], hA);
        atomicAdd(&xacc[(size_t)nB * DD + col], hB);
      }
    }
  }
}

// ---------------- fused symmetric edge MLP: e1=(src,dst), e2=(dst,src), combine + side loss ----------------
__global__ __launch_bounds__(128)
void kedge(const float* __restrict__ x, const int* __restrict__ srcI,
           const int* __restrict__ dstI, const __bf16* __restrict__ w1T,
           const float* __restrict__ b1, const __bf16* __restrict__ w2T,
           const float* __restrict__ b2, float* __restrict__ out,
           float* __restrict__ lossacc) {
  __shared__ __bf16 hid1[4][16][HW];
  __shared__ __bf16 hid2[4][16][HW];
  const int wave = threadIdx.x >> 5;
  const int lane = threadIdx.x & 31;
  const int hh   = lane >> 4;
  const int m16  = lane & 15;
  const int tile = blockIdx.x * 4 + wave;
  const int nTiles = EE / 16;
  const bool active = tile < nTiles;

  if (active) {
    int r = tile * 16 + m16;
    int s = srcI[r], d = dstI[r];
    const float* xs = x + (size_t)s * DD;
    const float* xd = x + (size_t)d * DD;
    v16bf af1[4], af2[4];
#pragma unroll
    for (int st = 0; st < 4; ++st)
#pragma unroll
      for (int v = 0; v < 8; ++v) {
        int kk = 32 * st + kpatA(v) + 8 * hh;
        int kb = (kk < DD) ? kk : kk - DD;
        float2 fs = *(const float2*)(xs + kb);
        float2 fd = *(const float2*)(xd + kb);
        float2 a = (kk < DD) ? fs : fd;   // e1 row = [x_src, x_dst]
        float2 b = (kk < DD) ? fd : fs;   // e2 row = [x_dst, x_src]
        af1[st][2 * v] = (__bf16)a.x; af1[st][2 * v + 1] = (__bf16)a.y;
        af2[st][2 * v] = (__bf16)b.x; af2[st][2 * v + 1] = (__bf16)b.y;
      }
    // shared GEMM1 for both orientations (B fragments reused)
#pragma unroll
    for (int ct = 0; ct < 8; ++ct) {
      int col = ct * 16 + m16;
      v8f acc1 = {0.f, 0.f, 0.f, 0.f, 0.f, 0.f, 0.f, 0.f};
      v8f acc2 = {0.f, 0.f, 0.f, 0.f, 0.f, 0.f, 0.f, 0.f};
      const __bf16* wrow = w1T + (size_t)col * HW;
#pragma unroll
      for (int st = 0; st < 4; ++st) {
        v16bf bfrag = load_bfrag(wrow, st, hh);
        acc1 = __builtin_amdgcn_wmma_f32_16x16x32_bf16(false, af1[st], false, bfrag,
                                                       (short)0, acc1, false, false);
        acc2 = __builtin_amdgcn_wmma_f32_16x16x32_bf16(false, af2[st], false, bfrag,
                                                       (short)0, acc2, false, false);
      }
      float bias = b1[col];
#pragma unroll
      for (int v = 0; v < 8; ++v) {
        float h1 = acc1[v] + bias; h1 = h1 > 0.f ? h1 : 0.f;
        float h2 = acc2[v] + bias; h2 = h2 > 0.f ? h2 : 0.f;
        hid1[wave][v + 8 * hh][col] = (__bf16)h1;
        hid2[wave][v + 8 * hh][col] = (__bf16)h2;
      }
    }
  }
  __syncthreads();
  if (active) {
    v16bf a1[4], a2[4];
#pragma unroll
    for (int st = 0; st < 4; ++st)
#pragma unroll
      for (int v = 0; v < 8; ++v) {
        int kk = 32 * st + kpatA(v) + 8 * hh;
        a1[st][2 * v] = hid1[wave][m16][kk]; a1[st][2 * v + 1] = hid1[wave][m16][kk + 1];
        a2[st][2 * v] = hid2[wave][m16][kk]; a2[st][2 * v + 1] = hid2[wave][m16][kk + 1];
      }
    float ls = 0.f;
#pragma unroll
    for (int ct = 0; ct < 4; ++ct) {
      int col = ct * 16 + m16;
      v8f acc1 = {0.f, 0.f, 0.f, 0.f, 0.f, 0.f, 0.f, 0.f};
      v8f acc2 = {0.f, 0.f, 0.f, 0.f, 0.f, 0.f, 0.f, 0.f};
      const __bf16* wrow = w2T + (size_t)col * HW;
#pragma unroll
      for (int st = 0; st < 4; ++st) {
        v16bf bfrag = load_bfrag(wrow, st, hh);
        acc1 = __builtin_amdgcn_wmma_f32_16x16x32_bf16(false, a1[st], false, bfrag,
                                                       (short)0, acc1, false, false);
        acc2 = __builtin_amdgcn_wmma_f32_16x16x32_bf16(false, a2[st], false, bfrag,
                                                       (short)0, acc2, false, false);
      }
      float bias = b2[col];
#pragma unroll
      for (int v = 0; v < 8; ++v) {
        float e1 = acc1[v] + bias;   // no relu on final layer
        float e2 = acc2[v] + bias;
        int gr = tile * 16 + v + 8 * hh;
        out[(size_t)gr * DD + col] = 0.5f * (e1 + e2);
        float df = e1 - e2;
        ls += df * df;
      }
    }
    atomicAdd(lossacc, ls);
  }
}

__global__ void kfinal(const float* __restrict__ lossacc, float* __restrict__ out) {
  if (blockIdx.x == 0 && threadIdx.x == 0)
    out[(size_t)EE * DD] = lossacc[0] / (float)((size_t)EE * DD);
}

extern "C" void kernel_launch(void* const* d_in, const int* in_sizes, int n_in,
                              void* d_out, int out_size, void* d_ws, size_t ws_size,
                              hipStream_t stream) {
  const float* ef  = (const float*)d_in[0];
  const int*   ei  = (const int*)d_in[1];
  const float* nw1 = (const float*)d_in[2];
  const float* nb1 = (const float*)d_in[3];
  const float* nw2 = (const float*)d_in[4];
  const float* nb2 = (const float*)d_in[5];
  const float* ew1 = (const float*)d_in[6];
  const float* eb1 = (const float*)d_in[7];
  const float* ew2 = (const float*)d_in[8];
  const float* eb2 = (const float*)d_in[9];
  const int* srcI = ei;
  const int* dstI = ei + EE;
  float* out = (float*)d_out;

  char* w = (char*)d_ws;
  float* x       = (float*)w; w += (size_t)NN * DD * 4;
  float* xacc    = (float*)w; w += (size_t)NN * DD * 4;
  float* deg     = (float*)w; w += (size_t)NN * 4;
  float* lossacc = (float*)w; w += 256;
  __bf16* w1T  = (__bf16*)w; w += (size_t)3 * HW * HW * 2;   // 3 x [128][128]
  __bf16* w2T  = (__bf16*)w; w += (size_t)3 * DD * HW * 2;   // 3 x [64][128]
  __bf16* ew1T = (__bf16*)w; w += (size_t)HW * HW * 2;       // [128][128]
  __bf16* ew2T = (__bf16*)w; w += (size_t)DD * HW * 2;       // [64][128]

  hipMemsetAsync(xacc, 0, (size_t)NN * DD * 4, stream);
  hipMemsetAsync(deg, 0, (size_t)NN * 4, stream);
  hipMemsetAsync(lossacc, 0, 4, stream);

  // weight prep: transpose + bf16
  for (int i = 0; i < 3; ++i) {
    kprep_w<<<(HW * HW + 255) / 256, 256, 0, stream>>>(nw1 + (size_t)i * HW * HW,
                                                       w1T + (size_t)i * HW * HW, HW, HW);
    kprep_w<<<(HW * DD + 255) / 256, 256, 0, stream>>>(nw2 + (size_t)i * HW * DD,
                                                       w2T + (size_t)i * DD * HW, HW, DD);
  }
  kprep_w<<<(HW * HW + 255) / 256, 256, 0, stream>>>(ew1, ew1T, HW, HW);
  kprep_w<<<(HW * DD + 255) / 256, 256, 0, stream>>>(ew2, ew2T, HW, DD);

  // initial node features
  {
    long long tot = 2LL * EE * DD;
    kscatter<<<(int)((tot + 255) / 256), 256, 0, stream>>>(ef, srcI, dstI, xacc, deg);
    knorm<<<(NN * DD + 255) / 256, 256, 0, stream>>>(x, xacc, deg, 0);
  }

  // 3 fused conv layers (2 row-tiles per wave)
  const int pairTiles = (2 * EE) / 32;
  const int convBlocks = (pairTiles + 3) / 4;
  for (int i = 0; i < 3; ++i) {
    kconv<<<convBlocks, 128, 0, stream>>>(x, xacc, srcI, dstI,
                                          w1T + (size_t)i * HW * HW, nb1 + (size_t)i * HW,
                                          w2T + (size_t)i * DD * HW, nb2 + (size_t)i * DD);
    knorm<<<(NN * DD + 255) / 256, 256, 0, stream>>>(x, xacc, deg, 1);
  }

  // fused symmetric edge MLP + side loss
  const int eTiles = EE / 16;
  kedge<<<(eTiles + 3) / 4, 128, 0, stream>>>(x, srcI, dstI, ew1T, eb1, ew2T, eb2, out, lossacc);
  kfinal<<<1, 32, 0, stream>>>(lossacc, out);
}
